// Model_66632122630656
// MI455X (gfx1250) — matrix operand order repaired and checked by reading, not compiled, not run on hardware
//
#include <hip/hip_runtime.h>
#include <math.h>

// ---------------------------------------------------------------------------
// Model dims (from reference)
// ---------------------------------------------------------------------------
#define BB     16      // batch
#define LL     1024    // input seq len
#define LPP    96      // decoder steps
#define HH     512     // HID
#define VOC    4000    // VOCAB
#define INDIM  80

typedef __attribute__((ext_vector_type(16))) _Float16 v16h;
typedef __attribute__((ext_vector_type(8)))  float    v8f;

#define WMMA_F32_F16(a,b,c) \
  __builtin_amdgcn_wmma_f32_16x16x32_f16(false,(a),false,(b),(short)0,(c),false,false)

static __device__ __forceinline__ float sigm(float x) { return 1.f / (1.f + __expf(-x)); }

// ---------------------------------------------------------------------------
// WMMA fragment loaders, per CDNA5 ISA layout tables (CDNA5 ISA §7.12.2).
//  A 16x32 (MxK) f16: row = lane&15; elem j -> k = (j<8 ? h8+j : 16+h8+j-8)
//  B 32x16 (KxN) from W (N x K) row-major f16: n = lane&15; elem j -> k = (j<8 ? h8+j : 16+h8+j-8)
//  C/D 16x16 f32: n = lane&15; row r -> m = r + 8*(lane>>4)
// f16 variants: per lane two contiguous 8-half (16B) chunks -> b128 loads.
// ---------------------------------------------------------------------------
static __device__ __forceinline__ v16h load_a_f16(const _Float16* base, int ld) {
  int lane = threadIdx.x & 31;
  const _Float16* p = base + (size_t)(lane & 15) * ld + ((lane >> 4) << 3);
  v16h a;
#pragma unroll
  for (int j = 0; j < 8; ++j) a[j] = p[j];
#pragma unroll
  for (int j = 0; j < 8; ++j) a[8 + j] = p[16 + j];
  return a;
}

static __device__ __forceinline__ v16h load_b_f16(const _Float16* base, int ld) {
  int lane = threadIdx.x & 31;
  const _Float16* p = base + (size_t)(lane & 15) * ld + ((lane >> 4) << 3);
  v16h b;
#pragma unroll
  for (int j = 0; j < 8; ++j) b[j] = p[j];
#pragma unroll
  for (int j = 0; j < 8; ++j) b[8 + j] = p[16 + j];
  return b;
}

static __device__ __forceinline__ v16h load_a_f32cvt(const float* base, int ld) {
  int lane = threadIdx.x & 31;
  const float* p = base + (size_t)(lane & 15) * ld + ((lane >> 4) << 3);
  v16h a;
#pragma unroll
  for (int j = 0; j < 8; ++j) a[j] = (_Float16)p[j];
#pragma unroll
  for (int j = 0; j < 8; ++j) a[8 + j] = (_Float16)p[16 + j];
  return a;
}

// ---------------------------------------------------------------------------
// Weight preparation: f32 -> packed f16 (once per launch; weights ~30MB total)
// ---------------------------------------------------------------------------
__global__ __launch_bounds__(256)
void cast2d_k(const float* __restrict__ src, int srcLd,
              _Float16* __restrict__ dst, int dstLd, int cols) {
  int row = blockIdx.x;
  for (int j = threadIdx.x; j < dstLd; j += 256)
    dst[(size_t)row * dstLd + j] =
        (j < cols) ? (_Float16)src[(size_t)row * srcLd + j] : (_Float16)0.f;
}

// conv weight (Cout,Cin,Kw) -> transposed padded (Kw,Cout,CinP), zero-filled
__global__ void convw_cast_k(const float* __restrict__ src, _Float16* __restrict__ dst,
                             int Cout, int Cin, int CinP, int Kw) {
  int i = blockIdx.x * blockDim.x + threadIdx.x;
  int total = Kw * Cout * CinP;
  if (i >= total) return;
  int ci = i % CinP;
  int rest = i / CinP;
  int co = rest % Cout;
  int kw = rest / Cout;
  dst[i] = (ci < Cin) ? (_Float16)src[((size_t)co * Cin + ci) * Kw + kw]
                      : (_Float16)0.f;
}

// ---------------------------------------------------------------------------
// Generic GEMM: C[M,N] = A[M,K] @ W16[N,K]^T + b1[N] + b2[N]
// A f32 activations (rows m = b*T + t), W16 packed f16.
// Output: f32 (Cf) or f16 (Ch). If tmT > 0, the output row is permuted to
// TIME-MAJOR (t*16 + b) so the downstream sequential scan consumes one
// contiguous 16x4H block per step (prefetchable, L2-streaming friendly).
// ---------------------------------------------------------------------------
__global__ __launch_bounds__(128)
void gemm_xwt_k(const float* __restrict__ A, const _Float16* __restrict__ W,
                const float* __restrict__ b1, const float* __restrict__ b2,
                float* __restrict__ Cf, _Float16* __restrict__ Ch,
                int M, int N, int K, int tmT) {
  int lane = threadIdx.x & 31;
  int wave = threadIdx.x >> 5;
  int m0 = blockIdx.y * 16;
  int n0 = (blockIdx.x * 4 + wave) * 16;
  if (n0 >= N) return;                    // wave-uniform exit, EXEC stays full
  int n  = n0 + (lane & 15);
  int mb = (lane >> 4) << 3;
  float bias = (b1 ? b1[n] : 0.f) + (b2 ? b2[n] : 0.f);
  v8f acc;
#pragma unroll
  for (int r = 0; r < 8; ++r) acc[r] = bias;
  for (int k0 = 0; k0 < K; k0 += 32) {
    v16h a = load_a_f32cvt(A + (size_t)m0 * K + k0, K);
    v16h b = load_b_f16(W + (size_t)n0 * K + k0, K);
    acc = WMMA_F32_F16(a, b, acc);
  }
#pragma unroll
  for (int r = 0; r < 8; ++r) {
    int m = m0 + mb + r;
    size_t row = tmT ? ((size_t)(m % tmT) * 16 + (m / tmT)) : (size_t)m;
    if (Cf) Cf[row * N + n] = acc[r];
    else    Ch[row * N + n] = (_Float16)acc[r];
  }
}

// ---------------------------------------------------------------------------
// Conv1d (same padding) as K_w shifted WMMA GEMMs.
// X:(B,T,Cin) f32, Wt:(Kw,Cout,CinP) packed f16, Y:(B,T,Cout) f32.
// ---------------------------------------------------------------------------
__global__ __launch_bounds__(128)
void conv1d_wmma_k(const float* __restrict__ X, const _Float16* __restrict__ Wt,
                   const float* __restrict__ bias, float* __restrict__ Y,
                   int T, int Cin, int CinP, int Cout, int Kw) {
  int lane = threadIdx.x & 31;
  int wave = threadIdx.x >> 5;
  int m0 = blockIdx.y * 16;                 // rows = b*T + t ; T % 16 == 0
  int n0 = (blockIdx.x * 4 + wave) * 16;
  if (n0 >= Cout) return;
  int pad = (Kw - 1) / 2;
  int m  = m0 + (lane & 15);
  int bidx = m / T, t = m % T;
  int half = lane >> 4;
  int n  = n0 + (lane & 15);
  int mb = half << 3;
  float bv = bias[n];
  v8f acc;
#pragma unroll
  for (int r = 0; r < 8; ++r) acc[r] = bv;
  for (int kw = 0; kw < Kw; ++kw) {
    int tt = t + kw - pad;
    bool ok = (tt >= 0) && (tt < T);
    const float* arow = X + ((size_t)bidx * T + (ok ? tt : 0)) * Cin;
    for (int k0 = 0; k0 < CinP; k0 += 32) {
      v16h a;
#pragma unroll
      for (int j = 0; j < 8; ++j) {
        int k = k0 + half * 8 + j;
        a[j] = (ok && k < Cin) ? (_Float16)arow[k] : (_Float16)0.f;
      }
#pragma unroll
      for (int j = 0; j < 8; ++j) {
        int k = k0 + 16 + half * 8 + j;
        a[8 + j] = (ok && k < Cin) ? (_Float16)arow[k] : (_Float16)0.f;
      }
      v16h b = load_b_f16(Wt + ((size_t)kw * Cout + n0) * CinP + k0, CinP);
      acc = WMMA_F32_F16(a, b, acc);
    }
  }
#pragma unroll
  for (int r = 0; r < 8; ++r) Y[(size_t)(m0 + mb + r) * Cout + n] = acc[r];
}

// ---------------------------------------------------------------------------
// LayerNorm over channels + LeakyReLU(0.1), in-place. One block per (b,t) row.
// ---------------------------------------------------------------------------
__global__ __launch_bounds__(256)
void ln_leaky_k(float* __restrict__ Y, const float* __restrict__ g,
                const float* __restrict__ b, int C) {
  __shared__ float red[256];
  float* y = Y + (size_t)blockIdx.x * C;
  int tid = threadIdx.x;
  float s = 0.f, s2 = 0.f;
  for (int i = tid; i < C; i += 256) { float v = y[i]; s += v; s2 += v * v; }
  red[tid] = s; __syncthreads();
  for (int w = 128; w > 0; w >>= 1) { if (tid < w) red[tid] += red[tid + w]; __syncthreads(); }
  float mean = red[0] / (float)C; __syncthreads();
  red[tid] = s2; __syncthreads();
  for (int w = 128; w > 0; w >>= 1) { if (tid < w) red[tid] += red[tid + w]; __syncthreads(); }
  float var = red[0] / (float)C - mean * mean;
  float inv = rsqrtf(var + 1e-5f);
  for (int i = tid; i < C; i += 256) {
    float o = (y[i] - mean) * inv * g[i] + b[i];
    y[i] = (o >= 0.f) ? o : 0.1f * o;
  }
}

// ---------------------------------------------------------------------------
// Persistent LSTM scan. One 1024-thread workgroup per direction (32 wave32s on
// one WGP). c,g f32 + h f16 in LDS (352*H bytes; 176KB @ H=512 -- needs
// CDNA5's 320KB WGP LDS). pre is TIME-MAJOR f16 (T,16,4H): one contiguous
// 64KB block per step, prefetched one step ahead via global_prefetch_b8.
// Per step: g = pre16[t] + h @ Whh16^T (WMMA); Whh16 (<=2MB) L2-resident.
// ---------------------------------------------------------------------------
__global__ __launch_bounds__(1024)
void lstm_scan_k(const _Float16* __restrict__ pre_f, const _Float16* __restrict__ pre_b,
                 const _Float16* __restrict__ Whh_f, const _Float16* __restrict__ Whh_b,
                 float* __restrict__ out, int T, int H, int outStride, int bidir) {
  extern __shared__ float smem[];
  float* c = smem;                          // 16*H f32
  float* g = c + 16 * H;                    // 16*4H f32
  _Float16* h = (_Float16*)(g + 16 * 4 * H);// 16*H f16
  int dir = bidir ? blockIdx.x : 0;
  const _Float16* pre = dir ? pre_b : pre_f;
  const _Float16* Whh = dir ? Whh_b : Whh_f;
  int tid = threadIdx.x, lane = tid & 31, wave = tid >> 5;
  int N4 = 4 * H;
  int tilesPerWave = H >> 7;                // 2 @ H=256, 4 @ H=512
  for (int i = tid; i < 16 * H; i += 1024) { h[i] = (_Float16)0.f; c[i] = 0.f; }
  __syncthreads();
  const size_t stepBytes = (size_t)16 * N4 * sizeof(_Float16);
  for (int t = 0; t < T; ++t) {
    int tio = dir ? (T - 1 - t) : t;
    // prefetch next step's contiguous pre block (covers stepBytes with 64B/thread)
    if (t + 1 < T) {
      int nxt = dir ? (T - 2 - t) : (t + 1);
      const char* pf = (const char*)(pre + (size_t)nxt * 16 * N4);
      size_t off = (size_t)tid * 64;
      if (off < stepBytes) __builtin_prefetch(pf + off, 0, 0);
    }
    const _Float16* pre_t = pre + (size_t)tio * 16 * N4;   // (16, 4H) block
    for (int tt = 0; tt < tilesPerWave; ++tt) {
      int n0 = (wave * tilesPerWave + tt) * 16;
      int n  = n0 + (lane & 15);
      int mb = (lane >> 4) << 3;
      v8f acc;
#pragma unroll
      for (int r = 0; r < 8; ++r)
        acc[r] = (float)pre_t[(size_t)(mb + r) * N4 + n];
      for (int k0 = 0; k0 < H; k0 += 32) {
        v16h a = load_a_f16(h + k0, H);
        v16h b = load_b_f16(Whh + (size_t)n0 * H + k0, H);
        acc = WMMA_F32_F16(a, b, acc);
      }
#pragma unroll
      for (int r = 0; r < 8; ++r) g[(mb + r) * N4 + n] = acc[r];
    }
    __syncthreads();
    for (int i = tid; i < 16 * H; i += 1024) {
      int bi = i / H, j = i % H;
      float gi = g[bi * N4 + j];
      float gf = g[bi * N4 + H + j];
      float gg = g[bi * N4 + 2 * H + j];
      float go = g[bi * N4 + 3 * H + j];
      float cc = sigm(gf) * c[i] + sigm(gi) * tanhf(gg);
      float hv = sigm(go) * tanhf(cc);
      c[i] = cc; h[i] = (_Float16)hv;
      out[((size_t)bi * T + tio) * outStride + dir * H + j] = hv;
    }
    __syncthreads();
  }
}

// ---------------------------------------------------------------------------
// Time pooling: out(B,T2,C) = in(B,2*T2,C) pairs summed
// ---------------------------------------------------------------------------
__global__ void pool2_k(const float* __restrict__ in, float* __restrict__ out,
                        int T2, int C) {
  size_t i = (size_t)blockIdx.x * blockDim.x + threadIdx.x;
  size_t total = (size_t)BB * T2 * C;
  if (i >= total) return;
  int cch = (int)(i % C);
  size_t bt = i / C;
  int b = (int)(bt / T2), t = (int)(bt % T2);
  const float* p = in + ((size_t)b * (2 * T2) + 2 * t) * C + cch;
  out[i] = p[0] + p[C];
}

// ---------------------------------------------------------------------------
// preY (TIME-MAJOR (LPP,16,4H) f32):
//   preY[t,b,:] = d1.Wih[:, H + temp_y[b,t]] + bih + bhh
// ---------------------------------------------------------------------------
__global__ __launch_bounds__(256)
void build_preY_k(const int* __restrict__ y, const float* __restrict__ Wih,
                  const float* __restrict__ bih, const float* __restrict__ bhh,
                  float* __restrict__ preY) {
  int row = blockIdx.x;              // b*LPP + t
  int b = row / LPP, t = row % LPP;
  int tok = (t == 0) ? 0 : y[b * LPP + t - 1];
  const int ld = HH + VOC;
  float* dst = preY + ((size_t)t * 16 + b) * (4 * HH);
  for (int j = threadIdx.x; j < 4 * HH; j += 256)
    dst[j] = Wih[(size_t)j * ld + HH + tok] + bih[j] + bhh[j];
}

// ---------------------------------------------------------------------------
// Attention decoder (d1): single persistent workgroup, 96 steps.
// Per step: scores=softmax(h.x/sqrt(H)); ctx=score.x;
//           g = preY[t] + ctx@Wih16[:, :H]^T + h@Whh16^T   (WMMA)
// h/ctx f16 in LDS. preY time-major + prefetched one step ahead.
// ---------------------------------------------------------------------------
__global__ __launch_bounds__(1024)
void decoder_attn_k(const float* __restrict__ xenc, const float* __restrict__ preY,
                    const _Float16* __restrict__ Wih16, const _Float16* __restrict__ Whh16,
                    float* __restrict__ a1) {
  extern __shared__ float smem[];
  const int Lx = 128, N4 = 4 * HH;
  float* c   = smem;                          // 16*512 f32
  float* g   = c + 16 * HH;                   // 16*2048 f32
  float* s   = g + 16 * N4;                   // 16*128 f32
  _Float16* h   = (_Float16*)(s + 16 * Lx);   // 16*512 f16
  _Float16* ctx = h + 16 * HH;                // 16*512 f16
  int tid = threadIdx.x, lane = tid & 31, wave = tid >> 5;
  for (int i = tid; i < 16 * HH; i += 1024) { h[i] = (_Float16)0.f; c[i] = 0.f; }
  __syncthreads();
  const float rscale = rsqrtf((float)HH);
  const size_t stepBytes = (size_t)16 * N4 * sizeof(float);
  for (int t = 0; t < LPP; ++t) {
    if (t + 1 < LPP) {
      const char* pf = (const char*)(preY + (size_t)(t + 1) * 16 * N4);
      size_t off = (size_t)tid * 128;
      if (off < stepBytes) __builtin_prefetch(pf + off, 0, 0);
    }
    // attention scores (2048 dot-products of length 512)
    for (int i = tid; i < 16 * Lx; i += 1024) {
      int b = i / Lx, l = i % Lx;
      const float* xr = xenc + ((size_t)b * Lx + l) * HH;
      const _Float16* hr = h + b * HH;
      float d = 0.f;
      for (int k = 0; k < HH; ++k) d += (float)hr[k] * xr[k];
      s[i] = d * rscale;
    }
    __syncthreads();
    if (tid < 16) {
      float mx = -1e30f;
      for (int l = 0; l < Lx; ++l) mx = fmaxf(mx, s[tid * Lx + l]);
      float su = 0.f;
      for (int l = 0; l < Lx; ++l) { float e = __expf(s[tid * Lx + l] - mx); s[tid * Lx + l] = e; su += e; }
      float inv = 1.f / su;
      for (int l = 0; l < Lx; ++l) s[tid * Lx + l] *= inv;
    }
    __syncthreads();
    // context
    for (int i = tid; i < 16 * HH; i += 1024) {
      int b = i / HH, j = i % HH;
      const float* xb = xenc + (size_t)b * Lx * HH + j;
      const float* sb = s + b * Lx;
      float a = 0.f;
      for (int l = 0; l < Lx; ++l) a += sb[l] * xb[(size_t)l * HH];
      ctx[i] = (_Float16)a;
    }
    __syncthreads();
    // gate pre-activations via WMMA: 128 N-tiles over 32 waves
    const float* preY_t = preY + (size_t)t * 16 * N4;
    for (int tt = 0; tt < 4; ++tt) {
      int n0 = (wave * 4 + tt) * 16;
      int n  = n0 + (lane & 15);
      int mb = (lane >> 4) << 3;
      v8f acc;
#pragma unroll
      for (int r = 0; r < 8; ++r)
        acc[r] = preY_t[(size_t)(mb + r) * N4 + n];
      for (int k0 = 0; k0 < HH; k0 += 32) {        // ctx @ Wih[:, :H]^T
        v16h a = load_a_f16(ctx + k0, HH);
        v16h b = load_b_f16(Wih16 + (size_t)n0 * HH + k0, HH);
        acc = WMMA_F32_F16(a, b, acc);
      }
      for (int k0 = 0; k0 < HH; k0 += 32) {        // h @ Whh^T
        v16h a = load_a_f16(h + k0, HH);
        v16h b = load_b_f16(Whh16 + (size_t)n0 * HH + k0, HH);
        acc = WMMA_F32_F16(a, b, acc);
      }
#pragma unroll
      for (int r = 0; r < 8; ++r) g[(mb + r) * N4 + n] = acc[r];
    }
    __syncthreads();
    for (int i = tid; i < 16 * HH; i += 1024) {
      int b = i / HH, j = i % HH;
      float gi = g[b * N4 + j];
      float gf = g[b * N4 + HH + j];
      float gg = g[b * N4 + 2 * HH + j];
      float go = g[b * N4 + 3 * HH + j];
      float cc = sigm(gf) * c[i] + sigm(gi) * tanhf(gg);
      float hv = sigm(go) * tanhf(cc);
      c[i] = cc; h[i] = (_Float16)hv;
      a1[((size_t)b * LPP + t) * HH + j] = hv;
    }
    __syncthreads();
  }
}

// ---------------------------------------------------------------------------
// Softmax + NLL pieces
// ---------------------------------------------------------------------------
__global__ void zero_k(float* p, int n) {
  int i = blockIdx.x * blockDim.x + threadIdx.x;
  if (i < n) p[i] = 0.f;
}

__global__ __launch_bounds__(256)
void softmax_loss_k(const float* __restrict__ logits, const int* __restrict__ y,
                    const float* __restrict__ pad, float* __restrict__ acc) {
  __shared__ float red[256];
  int row = blockIdx.x;
  const float* lr = logits + (size_t)row * VOC;
  int tid = threadIdx.x;
  float m = -1e30f;
  for (int j = tid; j < VOC; j += 256) m = fmaxf(m, lr[j]);
  red[tid] = m; __syncthreads();
  for (int w = 128; w > 0; w >>= 1) { if (tid < w) red[tid] = fmaxf(red[tid], red[tid + w]); __syncthreads(); }
  m = red[0]; __syncthreads();
  float s = 0.f;
  for (int j = tid; j < VOC; j += 256) s += __expf(lr[j] - m);
  red[tid] = s; __syncthreads();
  for (int w = 128; w > 0; w >>= 1) { if (tid < w) red[tid] += red[tid + w]; __syncthreads(); }
  s = red[0];
  if (tid == 0) {
    float smv = __expf(lr[y[row]] - m) / s;
    float pf  = pad[row];
    atomicAdd(&acc[0], -logf(smv + 1e-8f) * pf);
    atomicAdd(&acc[1], pf);
  }
}

__global__ void finalize_loss_k(const float* acc, float* dst) {
  if (threadIdx.x == 0) dst[0] = acc[0] / (acc[1] + 1e-8f);
}

// ---------------------------------------------------------------------------
// Host orchestration
// ---------------------------------------------------------------------------
extern "C" void kernel_launch(void* const* d_in, const int* in_sizes, int n_in,
                              void* d_out, int out_size, void* d_ws, size_t ws_size,
                              hipStream_t stream) {
  (void)in_sizes; (void)n_in; (void)out_size; (void)ws_size;
  // ---- inputs, setup_inputs()/ _init_params() insertion order -------------
  int idx = 0;
  const float* x        = (const float*)d_in[idx++];   // (16,1024,80)
  const int*   y        = (const int*)  d_in[idx++];   // (16,96)
  const float* pad      = (const float*)d_in[idx++];   // (16,96)
  const float* cnn1_w   = (const float*)d_in[idx++];
  const float* cnn1_b   = (const float*)d_in[idx++];
  const float* ln1_g    = (const float*)d_in[idx++];
  const float* ln1_b    = (const float*)d_in[idx++];
  const float* l1f_Wih  = (const float*)d_in[idx++];
  const float* l1f_Whh  = (const float*)d_in[idx++];
  const float* l1f_bih  = (const float*)d_in[idx++];
  const float* l1f_bhh  = (const float*)d_in[idx++];
  const float* l1b_Wih  = (const float*)d_in[idx++];
  const float* l1b_Whh  = (const float*)d_in[idx++];
  const float* l1b_bih  = (const float*)d_in[idx++];
  const float* l1b_bhh  = (const float*)d_in[idx++];
  const float* cnn2_w   = (const float*)d_in[idx++];
  const float* cnn2_b   = (const float*)d_in[idx++];
  const float* ln2_g    = (const float*)d_in[idx++];
  const float* ln2_b    = (const float*)d_in[idx++];
  const float* l2f_Wih  = (const float*)d_in[idx++];
  const float* l2f_Whh  = (const float*)d_in[idx++];
  const float* l2f_bih  = (const float*)d_in[idx++];
  const float* l2f_bhh  = (const float*)d_in[idx++];
  const float* l2b_Wih  = (const float*)d_in[idx++];
  const float* l2b_Whh  = (const float*)d_in[idx++];
  const float* l2b_bih  = (const float*)d_in[idx++];
  const float* l2b_bhh  = (const float*)d_in[idx++];
  const float* cnn3_w   = (const float*)d_in[idx++];
  const float* cnn3_b   = (const float*)d_in[idx++];
  const float* ln3_g    = (const float*)d_in[idx++];
  const float* ln3_b    = (const float*)d_in[idx++];
  const float* l3f_Wih  = (const float*)d_in[idx++];
  const float* l3f_Whh  = (const float*)d_in[idx++];
  const float* l3f_bih  = (const float*)d_in[idx++];
  const float* l3f_bhh  = (const float*)d_in[idx++];
  const float* l3b_Wih  = (const float*)d_in[idx++];
  const float* l3b_Whh  = (const float*)d_in[idx++];
  const float* l3b_bih  = (const float*)d_in[idx++];
  const float* l3b_bhh  = (const float*)d_in[idx++];
  const float* d1_Wih   = (const float*)d_in[idx++];   // (2048, 4512)
  const float* d1_Whh   = (const float*)d_in[idx++];   // (2048, 512)
  const float* d1_bih   = (const float*)d_in[idx++];
  const float* d1_bhh   = (const float*)d_in[idx++];
  const float* d2_Wih   = (const float*)d_in[idx++];   // (2048, 512)
  const float* d2_Whh   = (const float*)d_in[idx++];   // (2048, 512)
  const float* d2_bih   = (const float*)d_in[idx++];
  const float* d2_bhh   = (const float*)d_in[idx++];
  const float* lin_w    = (const float*)d_in[idx++];   // (4000, 512)
  const float* lin_b    = (const float*)d_in[idx++];

  // ---- workspace layout (needs ~225MB) ------------------------------------
  char* ws = (char*)d_ws;
  float*     slotA = (float*)    (ws + ((size_t)0   << 20)); // 32MB act (f32)
  float*     slotB = (float*)    (ws + ((size_t)32  << 20)); // 32MB pooled/xenc
  _Float16*  preF  = (_Float16*) (ws + ((size_t)64  << 20)); // 32MB gate pre f16
  _Float16*  preB  = (_Float16*) (ws + ((size_t)96  << 20)); // 32MB gate pre f16
  float*     hsBuf = (float*)    (ws + ((size_t)128 << 20)); // 32MB BiLSTM concat
  float*     preY  = (float*)    (ws + ((size_t)160 << 20)); // 12.6MB f32
  _Float16*  pre2  = (_Float16*) (ws + ((size_t)176 << 20)); // 6.3MB f16
  float*     a1    = (float*)    (ws + ((size_t)184 << 20)); // 3MB
  float*     a2    = (float*)    (ws + ((size_t)188 << 20)); // 3MB
  float*     accum = (float*)    (ws + ((size_t)192 << 20)); // 2 floats
  size_t cur = (size_t)194 << 20;                            // f16 weight arena
  auto ALLOC_H = [&](size_t elems) {
    _Float16* p = (_Float16*)(ws + cur);
    cur += ((elems * sizeof(_Float16) + 255) & ~(size_t)255);
    return p;
  };
  float* logits  = (float*)d_out;
  float* lossOut = logits + (size_t)BB * LPP * VOC;

  // ---- cast weights to packed f16 ----------------------------------------
  auto CASTW = [&](const float* src, int rows, int cols) {          // packed NxK
    _Float16* dst = ALLOC_H((size_t)rows * cols);
    cast2d_k<<<rows, 256, 0, stream>>>(src, cols, dst, cols, cols);
    return dst;
  };
  auto CASTCONV = [&](const float* src, int Cout, int Cin, int CinP, int Kw) {
    _Float16* dst = ALLOC_H((size_t)Kw * Cout * CinP);
    int total = Kw * Cout * CinP;
    convw_cast_k<<<(total + 255) / 256, 256, 0, stream>>>(src, dst, Cout, Cin, CinP, Kw);
    return dst;
  };
  _Float16* w_c1   = CASTCONV(cnn1_w, 256, INDIM, 96, 7);
  _Float16* w_c2   = CASTCONV(cnn2_w, 512, 512, 512, 5);
  _Float16* w_c3   = CASTCONV(cnn3_w, 512, 1024, 1024, 3);
  _Float16* w_l1fW = CASTW(l1f_Wih, 1024, 256);
  _Float16* w_l1fU = CASTW(l1f_Whh, 1024, 256);
  _Float16* w_l1bW = CASTW(l1b_Wih, 1024, 256);
  _Float16* w_l1bU = CASTW(l1b_Whh, 1024, 256);
  _Float16* w_l2fW = CASTW(l2f_Wih, 2048, 512);
  _Float16* w_l2fU = CASTW(l2f_Whh, 2048, 512);
  _Float16* w_l2bW = CASTW(l2b_Wih, 2048, 512);
  _Float16* w_l2bU = CASTW(l2b_Whh, 2048, 512);
  _Float16* w_l3fW = CASTW(l3f_Wih, 1024, 512);
  _Float16* w_l3fU = CASTW(l3f_Whh, 1024, 256);
  _Float16* w_l3bW = CASTW(l3b_Wih, 1024, 512);
  _Float16* w_l3bU = CASTW(l3b_Whh, 1024, 256);
  _Float16* w_d1W  = ALLOC_H((size_t)2048 * 512);          // d1.Wih[:, :512]
  cast2d_k<<<2048, 256, 0, stream>>>(d1_Wih, HH + VOC, w_d1W, 512, 512);
  _Float16* w_d1U  = CASTW(d1_Whh, 2048, 512);
  _Float16* w_d2W  = CASTW(d2_Wih, 2048, 512);
  _Float16* w_d2U  = CASTW(d2_Whh, 2048, 512);
  _Float16* w_lin  = CASTW(lin_w, VOC, 512);

  // ---- launch helpers ------------------------------------------------------
  auto GEMM = [&](const float* A, const _Float16* W, const float* b1, const float* b2,
                  float* Cf, _Float16* Ch, int M, int N, int K, int tmT) {
    dim3 grid((N + 63) / 64, M / 16);
    gemm_xwt_k<<<grid, 128, 0, stream>>>(A, W, b1, b2, Cf, Ch, M, N, K, tmT);
  };
  auto CONV = [&](const float* X, const _Float16* Wt, const float* b, float* Y,
                  int T, int Cin, int CinP, int Cout, int Kw) {
    dim3 grid((Cout + 63) / 64, (BB * T) / 16);
    conv1d_wmma_k<<<grid, 128, 0, stream>>>(X, Wt, b, Y, T, Cin, CinP, Cout, Kw);
  };
  auto LN = [&](float* Y, const float* g, const float* b, int rows, int C) {
    ln_leaky_k<<<rows, 256, 0, stream>>>(Y, g, b, C);
  };
  auto LSTM = [&](const _Float16* pf, const _Float16* pb,
                  const _Float16* wf, const _Float16* wb,
                  float* out, int T, int H, int stride, int bidir) {
    size_t sm = (size_t)352 * H;              // c(64H) + g(256H) + h16(32H) bytes
    lstm_scan_k<<<bidir ? 2 : 1, 1024, sm, stream>>>(pf, pb, wf, wb, out, T, H, stride, bidir);
  };
  auto POOL = [&](const float* in, float* out, int T2, int C) {
    size_t total = (size_t)BB * T2 * C;
    pool2_k<<<(int)((total + 255) / 256), 256, 0, stream>>>(in, out, T2, C);
  };

  // ===== Encoder layer 1: T=1024, Cin=80 -> 256, H=256 =====================
  CONV(x, w_c1, cnn1_b, slotA, LL, INDIM, 96, 256, 7);
  LN(slotA, ln1_g, ln1_b, BB * LL, 256);
  GEMM(slotA, w_l1fW, l1f_bih, l1f_bhh, nullptr, preF, BB * LL, 1024, 256, LL);
  GEMM(slotA, w_l1bW, l1b_bih, l1b_bhh, nullptr, preB, BB * LL, 1024, 256, LL);
  LSTM(preF, preB, w_l1fU, w_l1bU, hsBuf, LL, 256, 512, 1);
  POOL(hsBuf, slotB, LL / 2, 512);                     // (16,512,512)

  // ===== Encoder layer 2: T=512, Cin=512 -> 512, H=512 =====================
  CONV(slotB, w_c2, cnn2_b, slotA, 512, 512, 512, 512, 5);
  LN(slotA, ln2_g, ln2_b, BB * 512, 512);
  GEMM(slotA, w_l2fW, l2f_bih, l2f_bhh, nullptr, preF, BB * 512, 2048, 512, 512);
  GEMM(slotA, w_l2bW, l2b_bih, l2b_bhh, nullptr, preB, BB * 512, 2048, 512, 512);
  LSTM(preF, preB, w_l2fU, w_l2bU, hsBuf, 512, 512, 1024, 1);
  POOL(hsBuf, slotB, 256, 1024);                       // (16,256,1024)

  // ===== Encoder layer 3: T=256, Cin=1024 -> 512, H=256 ====================
  CONV(slotB, w_c3, cnn3_b, slotA, 256, 1024, 1024, 512, 3);
  LN(slotA, ln3_g, ln3_b, BB * 256, 512);
  GEMM(slotA, w_l3fW, l3f_bih, l3f_bhh, nullptr, preF, BB * 256, 1024, 512, 256);
  GEMM(slotA, w_l3bW, l3b_bih, l3b_bhh, nullptr, preB, BB * 256, 1024, 512, 256);
  LSTM(preF, preB, w_l3fU, w_l3bU, hsBuf, 256, 256, 512, 1);
  POOL(hsBuf, slotB, 128, 512);                        // xenc (16,128,512)

  // ===== Decoder ===========================================================
  build_preY_k<<<BB * LPP, 256, 0, stream>>>(y, d1_Wih, d1_bih, d1_bhh, preY);
  {
    // c(32KB) + g(128KB) + s(8KB) + h16(16KB) + ctx16(16KB) = 200KB LDS
    size_t sm = (size_t)(16 * HH + 16 * 4 * HH + 16 * 128) * sizeof(float)
              + (size_t)(16 * HH * 2) * sizeof(_Float16);
    decoder_attn_k<<<1, 1024, sm, stream>>>(slotB, preY, w_d1W, w_d1U, a1);
  }
  GEMM(a1, w_d2W, d2_bih, d2_bhh, nullptr, pre2, BB * LPP, 2048, 512, LPP);
  LSTM(pre2, pre2, w_d2U, w_d2U, a2, LPP, 512, 512, 0);

  // ===== Vocab projection + loss ==========================================
  GEMM(a2, w_lin, lin_b, nullptr, logits, nullptr, BB * LPP, VOC, 512, 0);
  zero_k<<<1, 32, 0, stream>>>(accum, 2);
  softmax_loss_k<<<BB * LPP, 256, 0, stream>>>(logits, y, pad, accum);
  finalize_loss_k<<<1, 32, 0, stream>>>(accum, lossOut);
}
